// MSAAttention_70909910057162
// MI455X (gfx1250) — matrix operand order repaired and checked by reading, not compiled, not run on hardware
//
#include <hip/hip_runtime.h>

// ---------------------------------------------------------------- constants
#define S_DIM 256
#define R_DIM 256
#define CM    256
#define CZ    128
#define H_DIM 8
#define D_DIM 32
#define HD    256   // H*D

typedef __attribute__((ext_vector_type(16))) _Float16 v16h;
typedef __attribute__((ext_vector_type(8)))  _Float16 v8h;
typedef __attribute__((ext_vector_type(8)))  float    v8f;
typedef __attribute__((ext_vector_type(4)))  unsigned int u32x4;
typedef __attribute__((ext_vector_type(8)))  int      i32x8;
typedef __attribute__((ext_vector_type(4)))  int      i32x4;

#define WMMA_F16(A, B, C) \
  __builtin_amdgcn_wmma_f32_16x16x32_f16(false, (A), false, (B), (short)0, (C), false, false)

// --------------------------------------------------------- workspace layout
#define SZ_SRC 33554432ull                       // 65536*256 f16 bytes
constexpr size_t WS_MN    = 0;                   // LN(m) in f16        [65536][256]
constexpr size_t WS_ZB    = WS_MN    + SZ_SRC;   // pair bias f32       [H][R][R]
constexpr size_t WS_WQKVG = WS_ZB    + 8ull*256*256*4;  // packed W^T f16 [1024][256]
constexpr size_t WS_WO    = WS_WQKVG + 1024ull*256*2;   // packed Wo^T f16 [256][256]
constexpr size_t WS_Q     = WS_WO    + 256ull*256*2;    // q f16 [S][H][R][D]
constexpr size_t WS_K     = WS_Q     + SZ_SRC;
constexpr size_t WS_V     = WS_K     + SZ_SRC;
constexpr size_t WS_G     = WS_V     + SZ_SRC;   // sigmoid gate f16 [S][R][HD]
constexpr size_t WS_OG    = WS_G     + SZ_SRC;   // gated attn out f16 [S][R][HD]

// ------------------------------------------------ TDM 2D tile load to LDS
__device__ __forceinline__ void tdm_load_tile_f16(
    _Float16* lds_dst, const _Float16* gsrc,
    unsigned tile_d0, unsigned tile_d1,
    unsigned tensor_d0, unsigned tensor_d1, unsigned stride0)
{
#if __has_builtin(__builtin_amdgcn_tensor_load_to_lds)
  unsigned lds_off = (unsigned)(unsigned long long)(void*)lds_dst; // LDS aperture: low 32 bits = offset
  unsigned long long ga = (unsigned long long)gsrc;
  u32x4 g0;
  g0[0] = 1u;                                            // count=1, user descriptor
  g0[1] = lds_off;                                       // lds_addr
  g0[2] = (unsigned)ga;                                  // global_addr[31:0]
  g0[3] = (unsigned)((ga >> 32) & 0x01ffffffull) | (2u << 30);  // addr[56:32] | type=2
  i32x8 g1;
  g1[0] = (int)(1u << 16);                               // data_size=1 (2B), no multicast
  g1[1] = (int)((tensor_d0 & 0xffffu) << 16);            // tensor_dim0[15:0]
  g1[2] = (int)((tensor_d0 >> 16) | ((tensor_d1 & 0xffffu) << 16));
  g1[3] = (int)((tensor_d1 >> 16) | (tile_d0 << 16));    // tile_dim0
  g1[4] = (int)(tile_d1 & 0xffffu);                      // tile_dim1, tile_dim2=0
  g1[5] = (int)stride0;                                  // tensor_dim0_stride[31:0]
  g1[6] = 0;
  g1[7] = 0;
  i32x4 z4 = {0, 0, 0, 0};
#if defined(__clang_major__) && (__clang_major__ >= 23)
  i32x8 z8 = {0, 0, 0, 0, 0, 0, 0, 0};
  __builtin_amdgcn_tensor_load_to_lds(g0, g1, z4, z4, z8, 0);
#else
  __builtin_amdgcn_tensor_load_to_lds(g0, g1, z4, z4, 0);
#endif
#else
  for (unsigned r = 0; r < tile_d1; ++r)
    for (unsigned c = threadIdx.x & 31u; c < tile_d0; c += 32u)
      lds_dst[(size_t)r * tile_d0 + c] = gsrc[(size_t)r * stride0 + c];
#endif
}

// ---------------------------------------------- WMMA fragment load helpers
// A 16x32 f16, order used here: lane = row, lanes 0-15 hold K 0..15, lanes 16-31 hold K 16..31
__device__ __forceinline__ v16h load_a_frag(const _Float16* row, int kk, int hf) {
  return *(const v16h*)(row + kk + 16 * hf);
}
// B 32x16 f16, order used here: lane = column, lanes 0-15 hold K 0..15, lanes 16-31 hold K 16..31
__device__ __forceinline__ v16h load_b_frag(const _Float16* row, int kk, int hf) {
  return *(const v16h*)(row + kk + 16 * hf);
}

// =================================================================== pack W
// WqkvgT[n][c]  (n = proj*256 + h*32 + d), q pre-scaled by 1/sqrt(D); WoT[n][c]
__global__ __launch_bounds__(256) void packw_kernel(
    const float* __restrict__ wq, const float* __restrict__ wk,
    const float* __restrict__ wv, const float* __restrict__ wg,
    const float* __restrict__ wo, _Float16* __restrict__ wqkvgT,
    _Float16* __restrict__ woT)
{
  int idx = blockIdx.x * 256 + threadIdx.x;
  if (idx < 1024 * 256) {
    int n = idx >> 8, c = idx & 255;
    int proj = n >> 8, hd = n & 255;
    const float* src = proj == 0 ? wq : proj == 1 ? wk : proj == 2 ? wv : wg;
    float v = src[c * 256 + hd];
    if (proj == 0) v *= 0.17677669529663687f;   // 1/sqrt(32)
    wqkvgT[(size_t)n * 256 + c] = (_Float16)v;
  } else {
    int i2 = idx - 1024 * 256;                  // [n][c] over 256x256
    int n = i2 >> 8, c = i2 & 255;
    woT[i2] = (_Float16)wo[c * 256 + n];
  }
}

// ================================================== pair bias: LN(z) @ w_z
// zb[h][q][k] = rs*(A_h - mu*B_h) + C_h with A_h=sum z*lnw*wz, B_h=sum lnw*wz, C_h=sum lnb*wz
__global__ __launch_bounds__(256) void zbias_kernel(
    const float* __restrict__ z, const float* __restrict__ lnw,
    const float* __restrict__ lnb, const float* __restrict__ wz,
    float* __restrict__ zb)
{
  __shared__ float swz[CZ * H_DIM];
  __shared__ float slw[CZ], slb[CZ];
  __shared__ float sB[H_DIM], sC[H_DIM];
  int tid = threadIdx.x;
  for (int i = tid; i < CZ * H_DIM; i += 256) swz[i] = wz[i];
  if (tid < CZ) { slw[tid] = lnw[tid]; slb[tid] = lnb[tid]; }
  __syncthreads();
  if (tid < H_DIM) {
    float B = 0.f, C = 0.f;
    for (int c = 0; c < CZ; ++c) {
      B += slw[c] * swz[c * H_DIM + tid];
      C += slb[c] * swz[c * H_DIM + tid];
    }
    sB[tid] = B; sC[tid] = C;
  }
  __syncthreads();
  int qi = blockIdx.x, ki = tid;
  const float* zr = z + ((size_t)qi * R_DIM + ki) * CZ;
  float s1 = 0.f, s2 = 0.f, A[H_DIM] = {0.f};
  for (int c = 0; c < CZ; ++c) {
    float zc = zr[c];
    s1 += zc; s2 += zc * zc;
    float t = zc * slw[c];
#pragma unroll
    for (int hh = 0; hh < H_DIM; ++hh) A[hh] += t * swz[c * H_DIM + hh];
  }
  float mu = s1 * (1.f / CZ);
  float var = s2 * (1.f / CZ) - mu * mu;
  float rs = rsqrtf(var + 1e-5f);
#pragma unroll
  for (int hh = 0; hh < H_DIM; ++hh)
    zb[((size_t)hh * R_DIM + qi) * R_DIM + ki] = rs * (A[hh] - mu * sB[hh]) + sC[hh];
}

// ======================================================= LN(m) -> f16 rows
__global__ __launch_bounds__(256) void lnm_kernel(
    const float* __restrict__ m, const float* __restrict__ lnw,
    const float* __restrict__ lnb, _Float16* __restrict__ mn)
{
  int w = threadIdx.x >> 5, lane = threadIdx.x & 31;
  size_t row = (size_t)blockIdx.x * 8 + w;
  const float* x = m + row * CM;
  float xs[8], s1 = 0.f, s2 = 0.f;
#pragma unroll
  for (int j = 0; j < 8; ++j) {
    xs[j] = x[lane + 32 * j];
    s1 += xs[j]; s2 += xs[j] * xs[j];
  }
#pragma unroll
  for (int off = 16; off >= 1; off >>= 1) {
    s1 += __shfl_xor(s1, off, 32);
    s2 += __shfl_xor(s2, off, 32);
  }
  float mu = s1 * (1.f / CM);
  float rs = rsqrtf(s2 * (1.f / CM) - mu * mu + 1e-5f);
#pragma unroll
  for (int j = 0; j < 8; ++j) {
    int c = lane + 32 * j;
    mn[row * CM + c] = (_Float16)((xs[j] - mu) * rs * lnw[c] + lnb[c]);
  }
}

// ================================== fused QKVG projection GEMM (f16 WMMA)
// C[65536 x 1024] = mn[65536 x 256] * WqkvgT^T ; epilogue scatters q/k/v and sigmoid(g)
__global__ __launch_bounds__(256) void gemm_qkvg_kernel(
    const _Float16* __restrict__ mn, const _Float16* __restrict__ wT,
    const float* __restrict__ b_g, _Float16* __restrict__ qb,
    _Float16* __restrict__ kb, _Float16* __restrict__ vb,
    _Float16* __restrict__ gb)
{
  __shared__ _Float16 As[64][256];
  __shared__ _Float16 Bs[64][256];
  const int m0 = blockIdx.x * 64, n0 = blockIdx.y * 64;
  const int tid = threadIdx.x, w = tid >> 5, l = tid & 31;
  const int hf = l >> 4, nl = l & 15;
  if (w == 0) {
    tdm_load_tile_f16(&As[0][0], mn + (size_t)m0 * 256, 256, 64, 256, 65536, 256);
    tdm_load_tile_f16(&Bs[0][0], wT + (size_t)n0 * 256, 256, 64, 256, 1024, 256);
    __builtin_amdgcn_s_wait_tensorcnt(0);
  }
  __syncthreads();

  const int mstrip = w & 3;              // 4 strips of 16 rows
  const int nhalf  = (w >> 2) * 32;      // 0 or 32
  const _Float16* arow  = &As[mstrip * 16 + nl][0];
  const _Float16* brow0 = &Bs[nhalf + nl][0];
  const _Float16* brow1 = &Bs[nhalf + 16 + nl][0];
  v8f acc[2] = {{0,0,0,0,0,0,0,0},{0,0,0,0,0,0,0,0}};
#pragma unroll
  for (int kk = 0; kk < 256; kk += 32) {
    v16h a  = load_a_frag(arow, kk, hf);
    v16h b0 = load_b_frag(brow0, kk, hf);
    v16h b1 = load_b_frag(brow1, kk, hf);
    acc[0] = WMMA_F16(a, b0, acc[0]);
    acc[1] = WMMA_F16(a, b1, acc[1]);
  }
#pragma unroll
  for (int nt = 0; nt < 2; ++nt) {
    int gn = n0 + nhalf + nt * 16 + nl;
    int proj = gn >> 8, hd = gn & 255, hh = hd >> 5, dd = hd & 31;
#pragma unroll
    for (int e = 0; e < 8; ++e) {
      int gm = m0 + mstrip * 16 + e + 8 * hf;
      int ss = gm >> 8, rr = gm & 255;
      float v = acc[nt][e];
      size_t sidx = ((size_t)((ss * H_DIM + hh) * R_DIM + rr)) * D_DIM + dd;
      if (proj == 0)      qb[sidx] = (_Float16)v;
      else if (proj == 1) kb[sidx] = (_Float16)v;
      else if (proj == 2) vb[sidx] = (_Float16)v;
      else {
        float gv = v + b_g[hd];
        gb[(size_t)gm * HD + hd] = (_Float16)(1.f / (1.f + __expf(-gv)));
      }
    }
  }
}

// =========================== attention: logits WMMA + softmax + PV WMMA
// one block per (s,h); 128 threads = 4 waves, each wave owns 64 query rows
__global__ __launch_bounds__(128) void attn_kernel(
    const _Float16* __restrict__ qb, const _Float16* __restrict__ kb,
    const _Float16* __restrict__ vb, const _Float16* __restrict__ gb,
    const float* __restrict__ zb, const float* __restrict__ mask,
    _Float16* __restrict__ og)
{
  __shared__ _Float16 Ks[R_DIM][D_DIM];    // k as [r][d]  (logits B frags)
  __shared__ _Float16 VsT[D_DIM][R_DIM];   // v transposed (PV B frags)
  __shared__ _Float16 P[4][16][R_DIM];     // per-wave probability strip
  const int s = blockIdx.x >> 3, h = blockIdx.x & 7;
  const int tid = threadIdx.x, w = tid >> 5, l = tid & 31;
  const int hf = l >> 4, nl = l & 15;
  const size_t shbase = ((size_t)(s * H_DIM + h)) * R_DIM * D_DIM;

  if (w == 0)
    tdm_load_tile_f16(&Ks[0][0], kb + shbase, 8192, 1, 8192, 1, 8192);
  for (int i = tid; i < R_DIM * D_DIM; i += 128) {
    int r = i >> 5, d = i & 31;
    VsT[d][r] = vb[shbase + i];
  }
  if (w == 0) __builtin_amdgcn_s_wait_tensorcnt(0);
  __syncthreads();

  const float* maskrow = mask + (size_t)s * R_DIM;
  for (int st = 0; st < 4; ++st) {
    const int q0 = w * 64 + st * 16;
    const _Float16* qrow = qb + shbase + (size_t)(q0 + nl) * D_DIM;
    v16h aq = load_a_frag(qrow, 0, hf);
    v8f lg[16];
#pragma unroll
    for (int nt = 0; nt < 16; ++nt) {
      int key = nt * 16 + nl;
      float mbias = 1e9f * (maskrow[key] - 1.0f);
      const float* zrow = zb + ((size_t)(h * R_DIM + q0)) * R_DIM + key;
      v8f cini;
#pragma unroll
      for (int e = 0; e < 8; ++e) cini[e] = mbias + zrow[(size_t)(e + 8 * hf) * R_DIM];
      v16h bk = load_b_frag(&Ks[key][0], 0, hf);
      lg[nt] = WMMA_F16(aq, bk, cini);      // K=32 == D: one WMMA per tile
    }
    // ---- softmax across the 256 keys of each of this lane-group's rows
    float rinv[8];
#pragma unroll
    for (int e = 0; e < 8; ++e) {
      float mx = -3.0e38f;
#pragma unroll
      for (int nt = 0; nt < 16; ++nt) mx = fmaxf(mx, lg[nt][e]);
#pragma unroll
      for (int msk = 8; msk >= 1; msk >>= 1) mx = fmaxf(mx, __shfl_xor(mx, msk, 16));
      float sm = 0.f;
#pragma unroll
      for (int nt = 0; nt < 16; ++nt) { float p = __expf(lg[nt][e] - mx); lg[nt][e] = p; sm += p; }
#pragma unroll
      for (int msk = 8; msk >= 1; msk >>= 1) sm += __shfl_xor(sm, msk, 16);
      rinv[e] = 1.f / sm;
    }
#pragma unroll
    for (int nt = 0; nt < 16; ++nt)
#pragma unroll
      for (int e = 0; e < 8; ++e)
        P[w][e + 8 * hf][nt * 16 + nl] = (_Float16)(lg[nt][e] * rinv[e]);
    asm volatile("s_wait_dscnt 0" ::: "memory");   // wave-local LDS RAW guard

    // ---- o = P @ V  (K=256 in 8 chunks), fused gate multiply
    v8f oacc[2] = {{0,0,0,0,0,0,0,0},{0,0,0,0,0,0,0,0}};
    const _Float16* prow = &P[w][nl][0];
#pragma unroll
    for (int kc = 0; kc < 256; kc += 32) {
      v16h ap  = load_a_frag(prow, kc, hf);
      v16h bv0 = load_b_frag(&VsT[nl][0], kc, hf);
      v16h bv1 = load_b_frag(&VsT[16 + nl][0], kc, hf);
      oacc[0] = WMMA_F16(ap, bv0, oacc[0]);
      oacc[1] = WMMA_F16(ap, bv1, oacc[1]);
    }
#pragma unroll
    for (int dt = 0; dt < 2; ++dt) {
      int d = dt * 16 + nl, hd = h * D_DIM + d;
#pragma unroll
      for (int e = 0; e < 8; ++e) {
        int qr = q0 + e + 8 * hf;
        size_t o = ((size_t)(s * R_DIM + qr)) * HD + hd;
        og[o] = (_Float16)(oacc[dt][e] * (float)gb[o]);
      }
    }
  }
}

// ======================================== output projection GEMM (+ b_o)
__global__ __launch_bounds__(256) void gemm_out_kernel(
    const _Float16* __restrict__ og, const _Float16* __restrict__ woT,
    const float* __restrict__ b_o, float* __restrict__ out)
{
  __shared__ _Float16 As[64][256];
  __shared__ _Float16 Bs[64][256];
  const int m0 = blockIdx.x * 64, n0 = blockIdx.y * 64;
  const int tid = threadIdx.x, w = tid >> 5, l = tid & 31;
  const int hf = l >> 4, nl = l & 15;
  if (w == 0) {
    tdm_load_tile_f16(&As[0][0], og + (size_t)m0 * 256, 256, 64, 256, 65536, 256);
    tdm_load_tile_f16(&Bs[0][0], woT + (size_t)n0 * 256, 256, 64, 256, 256, 256);
    __builtin_amdgcn_s_wait_tensorcnt(0);
  }
  __syncthreads();

  const int mstrip = w & 3, nhalf = (w >> 2) * 32;
  const _Float16* arow  = &As[mstrip * 16 + nl][0];
  const _Float16* brow0 = &Bs[nhalf + nl][0];
  const _Float16* brow1 = &Bs[nhalf + 16 + nl][0];
  v8f acc[2] = {{0,0,0,0,0,0,0,0},{0,0,0,0,0,0,0,0}};
#pragma unroll
  for (int kk = 0; kk < 256; kk += 32) {
    v16h a  = load_a_frag(arow, kk, hf);
    v16h b0 = load_b_frag(brow0, kk, hf);
    v16h b1 = load_b_frag(brow1, kk, hf);
    acc[0] = WMMA_F16(a, b0, acc[0]);
    acc[1] = WMMA_F16(a, b1, acc[1]);
  }
#pragma unroll
  for (int nt = 0; nt < 2; ++nt) {
    int gn = n0 + nhalf + nt * 16 + nl;
    float bias = b_o[gn];
#pragma unroll
    for (int e = 0; e < 8; ++e) {
      int gm = m0 + mstrip * 16 + e + 8 * hf;
      out[(size_t)gm * CM + gn] = acc[nt][e] + bias;
    }
  }
}

// =================================================================== launch
extern "C" void kernel_launch(void* const* d_in, const int* in_sizes, int n_in,
                              void* d_out, int out_size, void* d_ws, size_t ws_size,
                              hipStream_t stream) {
  const float* m      = (const float*)d_in[0];
  const float* z      = (const float*)d_in[1];
  const float* mask   = (const float*)d_in[2];
  const float* ln_m_w = (const float*)d_in[3];
  const float* ln_m_b = (const float*)d_in[4];
  const float* ln_z_w = (const float*)d_in[5];
  const float* ln_z_b = (const float*)d_in[6];
  const float* w_z    = (const float*)d_in[7];
  const float* w_q    = (const float*)d_in[8];
  const float* w_k    = (const float*)d_in[9];
  const float* w_v    = (const float*)d_in[10];
  const float* w_g    = (const float*)d_in[11];
  const float* b_g    = (const float*)d_in[12];
  const float* w_o    = (const float*)d_in[13];
  const float* b_o    = (const float*)d_in[14];

  char* ws = (char*)d_ws;
  _Float16* mn     = (_Float16*)(ws + WS_MN);
  float*    zbp    = (float*)   (ws + WS_ZB);
  _Float16* wqkvgT = (_Float16*)(ws + WS_WQKVG);
  _Float16* woT    = (_Float16*)(ws + WS_WO);
  _Float16* qb     = (_Float16*)(ws + WS_Q);
  _Float16* kb     = (_Float16*)(ws + WS_K);
  _Float16* vb     = (_Float16*)(ws + WS_V);
  _Float16* gb     = (_Float16*)(ws + WS_G);
  _Float16* ogb    = (_Float16*)(ws + WS_OG);

  packw_kernel<<<1280, 256, 0, stream>>>(w_q, w_k, w_v, w_g, w_o, wqkvgT, woT);
  zbias_kernel<<<R_DIM, 256, 0, stream>>>(z, ln_z_w, ln_z_b, w_z, zbp);
  lnm_kernel<<<8192, 256, 0, stream>>>(m, ln_m_w, ln_m_b, mn);
  gemm_qkvg_kernel<<<dim3(1024, 16), 256, 0, stream>>>(mn, wqkvgT, b_g, qb, kb, vb, gb);
  attn_kernel<<<S_DIM * H_DIM, 128, 0, stream>>>(qb, kb, vb, gb, zbp, mask, ogb);
  gemm_out_kernel<<<dim3(1024, 4), 256, 0, stream>>>(ogb, woT, b_o, (float*)d_out);
}
